// LogSparseAttention_11622181503358
// MI455X (gfx1250) — matrix operand order repaired and checked by reading, not compiled, not run on hardware
//
#include <hip/hip_runtime.h>

// ---------------------------------------------------------------------------
// LogSparse attention for MI455X (gfx1250, wave32, WMMA).
// Dominant cost: two causal convs == GEMMs (M=4096, N=1024, K=16384) -> route
// through v_wmma_f32_16x16x32_f16 with a double-buffered LDS pipeline fed by
// GLOBAL_LOAD_ASYNC_TO_LDS_B128 (ASYNCcnt).  Attention epilogue -> f32 VALU.
// ---------------------------------------------------------------------------

#define B_  2
#define T_  2048
#define D_  1024
#define H_  16
#define W_  16
#define E_  8
#define HD_ 64
#define M_  (B_ * T_)

typedef __attribute__((ext_vector_type(4)))  float    v4f;
typedef __attribute__((ext_vector_type(8)))  float    v8f;
typedef __attribute__((ext_vector_type(8)))  _Float16 v8h;
typedef __attribute__((ext_vector_type(16))) _Float16 v16h;

static __device__ inline v16h combine16(v8h lo, v8h hi) {
    v16h r;
#pragma unroll
    for (int i = 0; i < 8; ++i) { r[i] = lo[i]; r[i + 8] = hi[i]; }
    return r;
}

// Async DMA: 16B global -> LDS, tracked by ASYNCcnt (CDNA5 path).
static __device__ inline void async_copy16(unsigned lds_addr, const void* gsrc) {
    asm volatile("global_load_async_to_lds_b128 %0, %1, off"
                 :: "v"(lds_addr), "v"(gsrc)
                 : "memory");
}
static __device__ inline unsigned lds_off(const void* p) {
    return (unsigned)(unsigned long long)p;  // flat LDS addr low 32 bits
}
static __device__ inline void wait_async_le8() {
    asm volatile("s_wait_asynccnt 0x8" ::: "memory");
}
static __device__ inline void wait_async_0() {
    asm volatile("s_wait_asynccnt 0x0" ::: "memory");
}

// -------------------- f32 -> f16 cast (x, v_w, p_w) ------------------------
__global__ void cast_f32_f16(const float* __restrict__ src,
                             _Float16* __restrict__ dst, int n) {
    int i = blockIdx.x * blockDim.x + threadIdx.x;
    if (i < n) dst[i] = (_Float16)src[i];
}

__global__ void zero_f16(_Float16* __restrict__ dst) {
    dst[threadIdx.x] = (_Float16)0.0f;
}

// ---------- conv weight pack: [O][I][W] f32 -> [O][W][I] f16 ---------------
__global__ __launch_bounds__(256) void pack_conv_w(const float* __restrict__ src,
                                                   _Float16* __restrict__ dst) {
    __shared__ _Float16 s[W_][64 + 8];
    const int o   = blockIdx.x;       // 0..D-1
    const int i0  = blockIdx.y * 64;  // i tile base
    const int tid = threadIdx.x;
    const float* sp = src + (size_t)o * D_ * W_ + (size_t)i0 * W_;
#pragma unroll
    for (int p = 0; p < 4; ++p) {
        int idx = tid + p * 256;          // contiguous reads
        int r   = idx >> 4;
        int w   = idx & 15;
        s[w][r] = (_Float16)sp[idx];
    }
    __syncthreads();
    _Float16* dp = dst + (size_t)o * W_ * D_;
#pragma unroll
    for (int p = 0; p < 4; ++p) {
        int idx = tid + p * 256;
        int w   = idx >> 6;
        int r   = idx & 63;
        dp[w * D_ + i0 + r] = s[w][r];    // contiguous writes
    }
}

// -------------------- WMMA GEMM / causal conv ------------------------------
// Y[r, o] = bias[o] + sum_w sum_i X[b, t + w - (numW-1), i] * Wt[o][w][i]
// numW=16 -> causal conv (q/k); numW=1 -> plain projection (v/p).
#define BM 128
#define BN 128
#define BK 64
#define NSTEP (D_ / BK)  // 16 k-stages per w

__global__ __launch_bounds__(256) void gemm_conv_wmma(
    const _Float16* __restrict__ X,   // [B*T][D]
    const _Float16* __restrict__ Wt,  // [O][numW][D]
    const float*    __restrict__ bias,
    float*          __restrict__ Y,   // [B*T][D]
    int numW,
    const _Float16* __restrict__ zbuf) {  // >=16B of zeros (causal pad src)
    __shared__ _Float16 As[2][BM][BK + 8];
    __shared__ _Float16 Bs[2][BN][BK + 8];

    const int tid   = threadIdx.x;
    const int lane  = tid & 31;
    const int wv    = tid >> 5;   // 0..7
    const int wm    = wv >> 2;    // 0..1  (M direction)
    const int wn    = wv & 3;     // 0..3  (N direction)
    const int tileM = blockIdx.y * BM;
    const int tileN = blockIdx.x * BN;

    // copy mapping: thread covers rows {crow+32p}, 16B segment cseg, p=0..3
    const int crow = tid >> 3;
    const int cseg = (tid & 7) * 8;

    const int lm    = lane & 15;
    const int abase = (lane < 16) ? 0 : 8;   // A: K = abase..+7, abase+16..+23
    const int bbase = (lane < 16) ? 0 : 8;   // B: K = bbase..+7, bbase+16..+23

    // Flattened k-iteration: addr advances by exactly BK halves per iter
    // (w -> w+1 adds D while i0 wraps by -(D-BK)).
    const _Float16* aptr[4];
    const _Float16* bptr[4];
    int tbase[4];
#pragma unroll
    for (int p = 0; p < 4; ++p) {
        int r    = tileM + crow + 32 * p;
        int b    = r >> 11;                // /T_
        int t0   = (r & (T_ - 1)) - (numW - 1);
        tbase[p] = t0;
        aptr[p]  = X + ((size_t)(b * T_) + t0) * D_ + cseg;
        int o    = tileN + crow + 32 * p;
        bptr[p]  = Wt + (size_t)o * numW * D_ + cseg;
    }

    v8f acc[4][2];
#pragma unroll
    for (int mi = 0; mi < 4; ++mi)
#pragma unroll
        for (int ni = 0; ni < 2; ++ni)
#pragma unroll
            for (int r = 0; r < 8; ++r) acc[mi][ni][r] = 0.0f;

    const int NIT = numW * NSTEP;

    auto issue = [&](int kiter, int st) {
        const int    w    = kiter >> 4;  // NSTEP == 16
        const size_t koff = (size_t)kiter * BK;
#pragma unroll
        for (int p = 0; p < 4; ++p) {
            int  row   = crow + 32 * p;
            bool valid = (tbase[p] + w) >= 0;
            const void* src = valid ? (const void*)(aptr[p] + koff)
                                    : (const void*)zbuf;
            async_copy16(lds_off(&As[st][row][cseg]), src);
        }
#pragma unroll
        for (int p = 0; p < 4; ++p) {
            int row = crow + 32 * p;
            async_copy16(lds_off(&Bs[st][row][cseg]), bptr[p] + koff);
        }
    };

    issue(0, 0);
    for (int it = 0; it < NIT; ++it) {
        const int st = it & 1;
        if (it + 1 < NIT) {
            issue(it + 1, st ^ 1);  // prefetch next stage into other buffer
            wait_async_le8();       // retire this stage's 8 copies (in-order)
        } else {
            wait_async_0();
        }
        __syncthreads();            // stage st visible to all waves

        // ---- compute: two 32-wide WMMA k-substeps over the 64-wide stage ----
#pragma unroll
        for (int ks = 0; ks < 2; ++ks) {
            const int kk = ks * 32;
            v16h afrag[4], bfrag[2];
#pragma unroll
            for (int mi = 0; mi < 4; ++mi) {
                int rm = wm * 64 + mi * 16 + lm;
                v8h lo = *(const v8h*)&As[st][rm][kk + abase];
                v8h hi = *(const v8h*)&As[st][rm][kk + abase + 16];
                afrag[mi] = combine16(lo, hi);
            }
#pragma unroll
            for (int ni = 0; ni < 2; ++ni) {
                int rn = wn * 32 + ni * 16 + lm;
                v8h lo = *(const v8h*)&Bs[st][rn][kk + bbase];
                v8h hi = *(const v8h*)&Bs[st][rn][kk + bbase + 16];
                bfrag[ni] = combine16(lo, hi);
            }
#pragma unroll
            for (int mi = 0; mi < 4; ++mi)
#pragma unroll
                for (int ni = 0; ni < 2; ++ni)
                    acc[mi][ni] = __builtin_amdgcn_wmma_f32_16x16x32_f16(
                        false, afrag[mi], false, bfrag[ni],
                        (short)0, acc[mi][ni], false, false);
        }
        __syncthreads();  // all waves done reading st before it+2 overwrites it
    }

    // ---- epilogue: VGPR r holds M=r (lanes 0-15) / M=r+8 (lanes 16-31) ----
    const int mrow_off = (lane < 16) ? 0 : 8;
#pragma unroll
    for (int mi = 0; mi < 4; ++mi)
#pragma unroll
        for (int ni = 0; ni < 2; ++ni) {
            int   gcol = tileN + wn * 32 + ni * 16 + lm;
            float bv   = bias[gcol];
#pragma unroll
            for (int r = 0; r < 8; ++r) {
                int grow = tileM + wm * 64 + mi * 16 + mrow_off + r;
                Y[(size_t)grow * D_ + gcol] = acc[mi][ni][r] + bv;
            }
        }
}

// --------------- log-shift raw scores: s[b,h,e,t] --------------------------
__global__ void shift_scores(const float* __restrict__ q,
                             const float* __restrict__ k,
                             float* __restrict__ s) {
    int idx = blockIdx.x * blockDim.x + threadIdx.x;  // B*H*E*T
    int t = idx % T_;
    int e = (idx / T_) % E_;
    int h = (idx / (T_ * E_)) % H_;
    int b = idx / (T_ * E_ * H_);
    int ts = (t + (1 << e)) & (T_ - 1);  // jnp.roll wrap
    const v4f* qp = (const v4f*)(q + (size_t)(b * T_ + t) * D_ + h * HD_);
    const v4f* kp = (const v4f*)(k + (size_t)(b * T_ + ts) * D_ + h * HD_);
    float sum = 0.0f;
#pragma unroll
    for (int i = 0; i < 16; ++i) {
        v4f a = qp[i], c = kp[i];
        sum += a[0] * c[0] + a[1] * c[1] + a[2] * c[2] + a[3] * c[3];
    }
    s[idx] = sum * 0.125f;  // 1/sqrt(HD)
}

// --------------- softmax over T per (b,h,e): one WG each -------------------
__global__ __launch_bounds__(256) void softmax_T(float* __restrict__ s) {
    __shared__ float red[256];
    const int base = blockIdx.x * T_;
    const int tid  = threadIdx.x;
    float vals[8];
    float m = -3.4e38f;
#pragma unroll
    for (int i = 0; i < 8; ++i) {
        vals[i] = s[base + tid + i * 256];
        m = fmaxf(m, vals[i]);
    }
    red[tid] = m; __syncthreads();
    for (int off = 128; off > 0; off >>= 1) {
        if (tid < off) red[tid] = fmaxf(red[tid], red[tid + off]);
        __syncthreads();
    }
    m = red[0]; __syncthreads();
    float sum = 0.0f;
#pragma unroll
    for (int i = 0; i < 8; ++i) { vals[i] = __expf(vals[i] - m); sum += vals[i]; }
    red[tid] = sum; __syncthreads();
    for (int off = 128; off > 0; off >>= 1) {
        if (tid < off) red[tid] += red[tid + off];
        __syncthreads();
    }
    float inv = 1.0f / red[0];
#pragma unroll
    for (int i = 0; i < 8; ++i) s[base + tid + i * 256] = vals[i] * inv;
}

// --------- fused local-window attention + log-shift combine ----------------
// One wave per (b,h,t) row; lane handles head dims {2*lane, 2*lane+1}.
__global__ __launch_bounds__(256) void attn_combine(
    const float* __restrict__ q, const float* __restrict__ k,
    const float* __restrict__ v, const float* __restrict__ alpha,
    _Float16* __restrict__ attn) {
    const int gw   = (blockIdx.x * blockDim.x + threadIdx.x) >> 5;
    const int lane = threadIdx.x & 31;
    const int t = gw % T_;
    const int h = (gw / T_) % H_;
    const int b = gw / (T_ * H_);
    const int d0 = lane * 2;
    const float scale = 0.125f;

    const float* qp = q + (size_t)(b * T_ + t) * D_ + h * HD_;
    const float q0 = qp[d0], q1 = qp[d0 + 1];

    float sc[W_];
#pragma unroll
    for (int w = 0; w < W_; ++w) {
        int ts = t + w - (W_ - 1);
        float p = 0.0f;
        if (ts >= 0) {
            const float* kp = k + (size_t)(b * T_ + ts) * D_ + h * HD_;
            p = q0 * kp[d0] + q1 * kp[d0 + 1];
        }
#pragma unroll
        for (int off = 16; off > 0; off >>= 1) p += __shfl_xor(p, off, 32);
        sc[w] = p * scale;  // padded slots give exact 0, matching reference
    }
    float m = sc[0];
#pragma unroll
    for (int w = 1; w < W_; ++w) m = fmaxf(m, sc[w]);
    float ssum = 0.0f;
#pragma unroll
    for (int w = 0; w < W_; ++w) { sc[w] = __expf(sc[w] - m); ssum += sc[w]; }
    const float inv = 1.0f / ssum;

    float o0 = 0.0f, o1 = 0.0f;
#pragma unroll
    for (int w = 0; w < W_; ++w) {
        int ts = t + w - (W_ - 1);
        if (ts >= 0) {
            const float* vp = v + (size_t)(b * T_ + ts) * D_ + h * HD_;
            float a = sc[w] * inv;
            o0 += a * vp[d0]; o1 += a * vp[d0 + 1];
        }
    }
#pragma unroll
    for (int e = 0; e < E_; ++e) {
        int ts = (t + (1 << e)) & (T_ - 1);
        float a = alpha[(((size_t)(b * H_ + h)) * E_ + e) * T_ + t];
        const float* vp = v + (size_t)(b * T_ + ts) * D_ + h * HD_;
        o0 += a * vp[d0]; o1 += a * vp[d0 + 1];
    }
    _Float16* op = attn + (size_t)(b * T_ + t) * D_ + h * HD_;
    op[d0] = (_Float16)o0; op[d0 + 1] = (_Float16)o1;
}

// ---------------------------------------------------------------------------
extern "C" void kernel_launch(void* const* d_in, const int* in_sizes, int n_in,
                              void* d_out, int out_size, void* d_ws, size_t ws_size,
                              hipStream_t stream) {
    (void)in_sizes; (void)n_in; (void)out_size; (void)ws_size;
    const float* x   = (const float*)d_in[0];
    const float* q_w = (const float*)d_in[1];
    const float* q_b = (const float*)d_in[2];
    const float* k_w = (const float*)d_in[3];
    const float* k_b = (const float*)d_in[4];
    const float* v_w = (const float*)d_in[5];
    const float* v_b = (const float*)d_in[6];
    const float* p_w = (const float*)d_in[7];
    const float* p_b = (const float*)d_in[8];
    float* out = (float*)d_out;

    char* ws = (char*)d_ws;
    size_t off = 0;
    auto salloc = [&](size_t bytes) {
        void* p = ws + off;
        off += (bytes + 255) & ~(size_t)255;
        return p;
    };
    _Float16* x_h    = (_Float16*)salloc((size_t)M_ * D_ * 2);       //  8 MB
    _Float16* attn_h = (_Float16*)salloc((size_t)M_ * D_ * 2);       //  8 MB
    _Float16* wq_h   = (_Float16*)salloc((size_t)D_ * D_ * W_ * 2);  // 32 MB
    _Float16* wk_h   = (_Float16*)salloc((size_t)D_ * D_ * W_ * 2);  // 32 MB
    _Float16* wv_h   = (_Float16*)salloc((size_t)D_ * D_ * 2);       //  2 MB
    _Float16* wp_h   = (_Float16*)salloc((size_t)D_ * D_ * 2);       //  2 MB
    float* qbuf  = (float*)salloc((size_t)M_ * D_ * 4);              // 16 MB
    float* kbuf  = (float*)salloc((size_t)M_ * D_ * 4);              // 16 MB
    float* vbuf  = (float*)salloc((size_t)M_ * D_ * 4);              // 16 MB
    float* alpha = (float*)salloc((size_t)B_ * H_ * E_ * T_ * 4);    //  2 MB
    _Float16* zbuf = (_Float16*)salloc(256);                         // pad src

    // 1) pack activations + weights to f16; zero pad-source slab
    {
        zero_f16<<<1, 128, 0, stream>>>(zbuf);
        int n = M_ * D_;
        cast_f32_f16<<<(n + 255) / 256, 256, 0, stream>>>(x, x_h, n);
        int nw = D_ * D_;
        cast_f32_f16<<<(nw + 255) / 256, 256, 0, stream>>>(v_w, wv_h, nw);
        cast_f32_f16<<<(nw + 255) / 256, 256, 0, stream>>>(p_w, wp_h, nw);
        dim3 pg(D_, D_ / 64);
        pack_conv_w<<<pg, 256, 0, stream>>>(q_w, wq_h);
        pack_conv_w<<<pg, 256, 0, stream>>>(k_w, wk_h);
    }
    // 2) q/k causal convs + v projection (WMMA GEMMs, async-LDS pipelined)
    dim3 gg(D_ / BN, M_ / BM);  // (8, 32)
    gemm_conv_wmma<<<gg, 256, 0, stream>>>(x_h, wq_h, q_b, qbuf, W_, zbuf);
    gemm_conv_wmma<<<gg, 256, 0, stream>>>(x_h, wk_h, k_b, kbuf, W_, zbuf);
    gemm_conv_wmma<<<gg, 256, 0, stream>>>(x_h, wv_h, v_b, vbuf, 1, zbuf);
    // 3) log-shift scores + softmax over T
    shift_scores<<<(B_ * H_ * E_ * T_) / 256, 256, 0, stream>>>(qbuf, kbuf, alpha);
    softmax_T<<<B_ * H_ * E_, 256, 0, stream>>>(alpha);
    // 4) fused local-window + shift combine -> f16
    attn_combine<<<(B_ * H_ * T_ * 32) / 256, 256, 0, stream>>>(qbuf, kbuf, vbuf,
                                                               alpha, attn_h);
    // 5) output projection (WMMA GEMM) -> f32 d_out
    gemm_conv_wmma<<<gg, 256, 0, stream>>>(attn_h, wp_h, p_b, out, 1, zbuf);
}
